// SwinTransformerBlock_8555574854491
// MI455X (gfx1250) — compile-verified, matrix operand order checked by reading, not run on hardware
//
#include <hip/hip_runtime.h>
#include <hip/hip_bf16.h>

// Swin block: B=32,H=W=112,C=96,heads=3,ws=7,shift=3 -> T=401408 tokens, 8192 windows
// HBM-bound (~97 GFLOP vs ~1.2GB traffic => ~50us floor @23.3TB/s), so:
//  - all GEMMs in f16-in / f32-acc WMMA (v_wmma_f32_16x16x32_f16), 64x32 tile/wave
//  - attention fully fused per (window,head) in LDS; relpos bias + shift mask + pad
//    precomputed into an additive table [4 mask cases][3 heads][64x64] that is fed
//    straight into the WMMA C accumulator (D = Q'K^T + table), zero epilogue ALU
//  - pointwise stages (LN1+shift+partition, unpartition+LN2+residual, GELU, bias)
//    fused into producers/consumers: every intermediate touches HBM once
//
// Workspace layout (needs 385,769,472 bytes):
//   [0, 221184)                 : f16 weights (qkv_w, proj_w, fc1_w, fc2_w)
//   [221184, 417792)            : attn additive table (4*3*64*64 f32)
//   B region [417792, +77MB)    : xw -> o -> h_ln2     (T*96 f16)
//   A region [+77MB, +308MB)    : qkv -> xr(f32) -> h1 (T*384 f16 max)

typedef __attribute__((ext_vector_type(16))) _Float16 v16h;
typedef __attribute__((ext_vector_type(8)))  float    v8f;

#define T_TOK 401408
#define NWIN  8192
#define QKSCALE 0.17677669529663687f   // 32^-0.5

union FragH { v16h v; int4 q[2]; };
union HalfQ { int4 q; _Float16 h[8]; };
union VF8   { v8f v; float f[8]; };
union H4    { int2 d; _Float16 h[4]; };

// One 16x32 f16 fragment in the CDNA5 WMMA A/B register layout.
// lane l: row = row0 + (l&15); group g = l>>4;
//   halves 0..7  <- base[row*ld + k0 + g*8 + 0..7]
//   halves 8..15 <- base[row*ld + k0 + 16 + g*8 + 0..7]
__device__ inline v16h load_frag(const _Float16* base, int ld, int row0, int k0) {
  const int lane = threadIdx.x & 31;
  const int g = lane >> 4, idx = lane & 15;
  const _Float16* p = base + (size_t)(row0 + idx) * ld + (size_t)(k0 + g * 8);
  FragH f;
  f.q[0] = *(const int4*)(p);
  f.q[1] = *(const int4*)(p + 16);
  return f.v;
}

__device__ inline v8f wmma16(v16h a, v16h b, v8f c) {
  return __builtin_amdgcn_wmma_f32_16x16x32_f16(false, a, false, b, (short)0, c,
                                                false, false);
}

__device__ inline float wave_sum(float v) {
#pragma unroll
  for (int off = 16; off > 0; off >>= 1) v += __shfl_xor(v, off, 32);
  return v;
}

__device__ inline int regid(int p) { return p < 105 ? 0 : (p < 109 ? 1 : 2); }

// ---------------------------------------------------------------- weights cvt
__global__ __launch_bounds__(256) void cvt_f16(const float* __restrict__ s,
                                               _Float16* __restrict__ d, int n) {
  int i = blockIdx.x * 256 + threadIdx.x;
  if (i < n) d[i] = (_Float16)s[i];
}

// ---------------------------- fused attention additive table (bias+mask+pad)
// tbl[cas][head][i*64+j]; cas = (wh==15)<<1 | (ww==15); pad entries = -1e9
__global__ __launch_bounds__(256)
void build_attn_table(const float* __restrict__ rpb, float* __restrict__ tbl) {
  int e = blockIdx.x * 256 + threadIdx.x;          // 4*3*4096 = 49152
  if (e >= 49152) return;
  int hc = e >> 12, idx = e & 4095;
  int cas = hc / 3, head = hc - cas * 3;
  int i = idx >> 6, j = idx & 63;
  float val = -1e9f;
  if (i < 49 && j < 49) {
    int ri = i / 7, ci = i - ri * 7;
    int rj = j / 7, cj = j - rj * 7;
    int rel = (ri - rj + 6) * 13 + (ci - cj + 6);
    int wh = (cas & 2) ? 15 : 0, ww = (cas & 1) ? 15 : 0;
    int regi = regid(wh * 7 + ri) * 3 + regid(ww * 7 + ci);
    int regj = regid(wh * 7 + rj) * 3 + regid(ww * 7 + cj);
    val = rpb[rel * 3 + head] + ((regi == regj) ? 0.f : -100.f);
  }
  tbl[e] = val;
}

// --------------------------------------- LN1 + cyclic shift + window partition
// one wave per window-layout token; lanes cover channels {l, l+32, l+64}
__global__ __launch_bounds__(256)
void ln1_window(const float* __restrict__ x, const float* __restrict__ gw,
                const float* __restrict__ gb, _Float16* __restrict__ xw) {
  int wave = blockIdx.x * 8 + (threadIdx.x >> 5);
  if (wave >= T_TOK) return;
  int lane = threadIdx.x & 31;
  int win = wave / 49, i = wave - win * 49;
  int b = win >> 8, wi = win & 255;
  int wh = wi >> 4, ww = wi & 15;
  int r = i / 7, c = i - r * 7;
  int h = wh * 7 + r + 3;  if (h >= 112) h -= 112;   // roll(-shift)
  int w = ww * 7 + c + 3;  if (w >= 112) w -= 112;
  const float* src = x + ((size_t)b * 12544 + (size_t)h * 112 + w) * 96;
  float v0 = src[lane], v1 = src[lane + 32], v2 = src[lane + 64];
  float mean = wave_sum(v0 + v1 + v2) * (1.f / 96.f);
  float d0 = v0 - mean, d1 = v1 - mean, d2 = v2 - mean;
  float var = wave_sum(d0 * d0 + d1 * d1 + d2 * d2) * (1.f / 96.f);
  float rs = rsqrtf(var + 1e-5f);
  _Float16* dst = xw + (size_t)wave * 96;
  dst[lane]      = (_Float16)(d0 * rs * gw[lane]      + gb[lane]);
  dst[lane + 32] = (_Float16)(d1 * rs * gw[lane + 32] + gb[lane + 32]);
  dst[lane + 64] = (_Float16)(d2 * rs * gw[lane + 64] + gb[lane + 64]);
}

// ---------------------------------------------------------------- WMMA GEMM
// out[M,N] = A[M,K] @ W[N,K]^T + bias ; 64x32 macro-tile per wave (4x2 WMMA)
// EPI: 0=store f16, 1=store f32, 2=GELU->f16, 3=accumulate into f32 out
template <int EPI>
__global__ __launch_bounds__(256)
void gemm_wmma(const _Float16* __restrict__ A, const _Float16* __restrict__ W,
               const float* __restrict__ bias, void* __restrict__ outv,
               int M, int N, int K) {
  const int wave = blockIdx.x * 8 + (threadIdx.x >> 5);
  const int n2 = N >> 5;
  const int tiles = (M >> 6) * n2;
  if (wave >= tiles) return;                     // wave-uniform: EXEC stays full
  const int tm = wave / n2, tn = wave - tm * n2;
  const int lane = threadIdx.x & 31;

  v8f acc[4][2] = {};
  for (int kc = 0; kc < K; kc += 32) {
    if (kc + 32 < K) {                           // prefetch next A K-chunk
      __builtin_prefetch(A + (size_t)(tm * 64 + lane) * K + kc + 32, 0, 3);
      __builtin_prefetch(A + (size_t)(tm * 64 + 32 + lane) * K + kc + 32, 0, 3);
    }
    v16h b0 = load_frag(W, K, tn * 32, kc);
    v16h b1 = load_frag(W, K, tn * 32 + 16, kc);
#pragma unroll
    for (int i = 0; i < 4; ++i) {
      v16h a = load_frag(A, K, tm * 64 + i * 16, kc);
      acc[i][0] = wmma16(a, b0, acc[i][0]);
      acc[i][1] = wmma16(a, b1, acc[i][1]);
    }
  }

  const int g = lane >> 4, idx = lane & 15;
#pragma unroll
  for (int j = 0; j < 2; ++j) {
    const int n = tn * 32 + j * 16 + idx;
    const float bb = bias ? bias[n] : 0.f;
#pragma unroll
    for (int i = 0; i < 4; ++i) {
#pragma unroll
      for (int r = 0; r < 8; ++r) {
        const int m = tm * 64 + i * 16 + g * 8 + r;
        float v = acc[i][j][r] + bb;
        const size_t o = (size_t)m * N + n;
        if (EPI == 0) {
          ((_Float16*)outv)[o] = (_Float16)v;
        } else if (EPI == 1) {
          ((float*)outv)[o] = v;
        } else if (EPI == 2) {
          float gl = 0.5f * v * (1.f + erff(v * 0.70710678118654752f));
          ((_Float16*)outv)[o] = (_Float16)gl;
        } else {
          float* po = (float*)outv;
          po[o] = po[o] + v;
        }
      }
    }
  }
}

// ------------------------------------------------- fused window attention
// one block (4 waves) per (window, head). N=49 padded to 64 in LDS.
__global__ __launch_bounds__(128)
void attn_kernel(const _Float16* __restrict__ qkv, _Float16* __restrict__ o,
                 const float* __restrict__ tbl) {
  __shared__ __align__(16) _Float16 Qs[64 * 32];   // pre-scaled by 32^-0.5
  __shared__ __align__(16) _Float16 Ks[64 * 32];
  __shared__ __align__(16) _Float16 Vt[32 * 64];   // transposed: Vt[d][j]
  __shared__ __align__(16) _Float16 Ps[64 * 64];
  __shared__ __align__(16) float Ss[64 * 64];

  const int win = blockIdx.x / 3, head = blockIdx.x - win * 3;
  const int wimg = win & 255;
  const int wh = wimg >> 4, ww = wimg & 15;
  const int cas = ((wh == 15) ? 2 : 0) | ((ww == 15) ? 1 : 0);
  const int tid = threadIdx.x;
  const size_t tokBase = (size_t)win * 49;
  const float* tb = tbl + ((size_t)(cas * 3 + head) << 12);  // L2-resident

  // stage Q (scaled), K, V(transposed); rows >= 49 zero. int4 = 8 halves/copy.
  for (int e = tid; e < 256; e += 128) {
    int i = e >> 2, q = e & 3;
    HalfQ qz, kz, vz;
    qz.q = int4{0, 0, 0, 0}; kz.q = qz.q; vz.q = qz.q;
    if (i < 49) {
      const _Float16* row = qkv + (tokBase + i) * 288 + head * 32;
      qz.q = ((const int4*)row)[q];
      kz.q = ((const int4*)(row + 96))[q];
      vz.q = ((const int4*)(row + 192))[q];
#pragma unroll
      for (int t = 0; t < 8; ++t) qz.h[t] = qz.h[t] * (_Float16)QKSCALE;
    }
    *(int4*)(Qs + i * 32 + q * 8) = qz.q;
    *(int4*)(Ks + i * 32 + q * 8) = kz.q;
#pragma unroll
    for (int t = 0; t < 8; ++t) Vt[(q * 8 + t) * 64 + i] = vz.h[t];
  }
  __syncthreads();

  const int wv = tid >> 5;
  const int lane = tid & 31, g = lane >> 4, idx = lane & 15;

  // S = Q'K^T + table via the WMMA C accumulator (table pre-loaded in D layout;
  // padded Q/K rows give A*B == 0 there and the table holds -1e9).
  {
    v16h a = load_frag(Qs, 32, wv * 16, 0);
#pragma unroll
    for (int tn = 0; tn < 4; ++tn) {
      v16h b = load_frag(Ks, 32, tn * 16, 0);
      VF8 c;
#pragma unroll
      for (int r = 0; r < 8; ++r)
        c.f[r] = tb[(wv * 16 + g * 8 + r) * 64 + tn * 16 + idx];
      c.v = wmma16(a, b, c.v);
#pragma unroll
      for (int r = 0; r < 8; ++r)
        Ss[(wv * 16 + g * 8 + r) * 64 + tn * 16 + idx] = c.f[r];
    }
  }
  __syncthreads();

  // row softmax (f32, float4-wide) -> P (f16, packed 4-half stores)
  if (tid < 64) {
    const float4* srow = (const float4*)(Ss + tid * 64);
    float mx = -1e30f;
#pragma unroll 4
    for (int j = 0; j < 16; ++j) {
      float4 t = srow[j];
      mx = fmaxf(mx, fmaxf(fmaxf(t.x, t.y), fmaxf(t.z, t.w)));
    }
    float sum = 0.f;
#pragma unroll 4
    for (int j = 0; j < 16; ++j) {
      float4 t = srow[j];
      sum += __expf(t.x - mx) + __expf(t.y - mx) + __expf(t.z - mx) + __expf(t.w - mx);
    }
    float inv = 1.f / sum;
    int2* prow = (int2*)(Ps + tid * 64);
#pragma unroll 4
    for (int j = 0; j < 16; ++j) {
      float4 t = srow[j];
      H4 p;
      p.h[0] = (_Float16)(__expf(t.x - mx) * inv);
      p.h[1] = (_Float16)(__expf(t.y - mx) * inv);
      p.h[2] = (_Float16)(__expf(t.z - mx) * inv);
      p.h[3] = (_Float16)(__expf(t.w - mx) * inv);
      prow[j] = p.d;
    }
  }
  __syncthreads();

  // O = P V : wave wv owns row-block wv, tn in {0,1} covers 32 head dims
#pragma unroll
  for (int tn = 0; tn < 2; ++tn) {
    v8f c = {};
#pragma unroll
    for (int kc = 0; kc < 64; kc += 32) {
      v16h a = load_frag(Ps, 64, wv * 16, kc);
      v16h b = load_frag(Vt, 64, tn * 16, kc);
      c = wmma16(a, b, c);
    }
#pragma unroll
    for (int r = 0; r < 8; ++r) {
      int i = wv * 16 + g * 8 + r;
      if (i < 49) {
        int d = tn * 16 + idx;
        o[(tokBase + i) * 96 + head * 32 + d] = (_Float16)c[r];
      }
    }
  }
}

// ------------------------ window reverse + reverse shift + residual + LN2
__global__ __launch_bounds__(256)
void unwindow_ln2(const float* __restrict__ xrw, const float* __restrict__ x0,
                  const float* __restrict__ gw, const float* __restrict__ gb,
                  float* __restrict__ out, _Float16* __restrict__ hln) {
  int wave = blockIdx.x * 8 + (threadIdx.x >> 5);
  if (wave >= T_TOK) return;
  int lane = threadIdx.x & 31;
  int b = wave / 12544, hw = wave - b * 12544;
  int h = hw / 112, w = hw - h * 112;
  int hs = h + 109; if (hs >= 112) hs -= 112;       // roll(+shift)
  int ws = w + 109; if (ws >= 112) ws -= 112;
  int wh = hs / 7, r = hs - wh * 7;
  int ww = ws / 7, c = ws - ww * 7;
  size_t twin = ((size_t)(b * 256 + wh * 16 + ww)) * 49 + r * 7 + c;
  const float* src = xrw + twin * 96;
  const float* sc  = x0 + (size_t)wave * 96;
  float v0 = src[lane], v1 = src[lane + 32], v2 = src[lane + 64];
  float* od = out + (size_t)wave * 96;
  od[lane]      = sc[lane]      + v0;               // shortcut + xr
  od[lane + 32] = sc[lane + 32] + v1;
  od[lane + 64] = sc[lane + 64] + v2;
  float mean = wave_sum(v0 + v1 + v2) * (1.f / 96.f);
  float d0 = v0 - mean, d1 = v1 - mean, d2 = v2 - mean;
  float var = wave_sum(d0 * d0 + d1 * d1 + d2 * d2) * (1.f / 96.f);
  float rs = rsqrtf(var + 1e-5f);
  _Float16* hd = hln + (size_t)wave * 96;
  hd[lane]      = (_Float16)(d0 * rs * gw[lane]      + gb[lane]);
  hd[lane + 32] = (_Float16)(d1 * rs * gw[lane + 32] + gb[lane + 32]);
  hd[lane + 64] = (_Float16)(d2 * rs * gw[lane + 64] + gb[lane + 64]);
}

// ---------------------------------------------------------------- launcher
extern "C" void kernel_launch(void* const* d_in, const int* in_sizes, int n_in,
                              void* d_out, int out_size, void* d_ws, size_t ws_size,
                              hipStream_t stream) {
  const float* x      = (const float*)d_in[0];
  const float* n1w    = (const float*)d_in[1];
  const float* n1b    = (const float*)d_in[2];
  const float* qkv_w  = (const float*)d_in[3];
  const float* qkv_b  = (const float*)d_in[4];
  const float* proj_w = (const float*)d_in[5];
  const float* proj_b = (const float*)d_in[6];
  const float* rpb    = (const float*)d_in[7];
  const float* n2w    = (const float*)d_in[8];
  const float* n2b    = (const float*)d_in[9];
  const float* fc1_w  = (const float*)d_in[10];
  const float* fc1_b  = (const float*)d_in[11];
  const float* fc2_w  = (const float*)d_in[12];
  const float* fc2_b  = (const float*)d_in[13];

  char* wsb = (char*)d_ws;
  _Float16* wQKV  = (_Float16*)wsb;         // 288*96
  _Float16* wPROJ = wQKV + 27648;           // 96*96
  _Float16* wFC1  = wPROJ + 9216;           // 384*96
  _Float16* wFC2  = wFC1 + 36864;           // 96*384
  float* attnTbl  = (float*)(wsb + 221184); // 4*3*64*64 f32 = 196608 B
  const size_t OFF_B = 221184 + 196608;                     // 417792
  const size_t OFF_A = OFF_B + (size_t)T_TOK * 96 * 2;
  _Float16* regB = (_Float16*)(wsb + OFF_B);
  char*     regA = wsb + OFF_A;

  _Float16* xw   = regB;                 // T x 96  f16
  _Float16* qkvb = (_Float16*)regA;      // T x 288 f16
  _Float16* obuf = regB;                 // T x 96  f16 (xw dead)
  float*    xrw  = (float*)regA;         // T x 96  f32 (qkv dead)
  _Float16* hln  = regB;                 // T x 96  f16 (o dead)
  _Float16* h1   = (_Float16*)regA;      // T x 384 f16 (xr dead)

  cvt_f16<<<(27648 + 255) / 256, 256, 0, stream>>>(qkv_w, wQKV, 27648);
  cvt_f16<<<(9216  + 255) / 256, 256, 0, stream>>>(proj_w, wPROJ, 9216);
  cvt_f16<<<(36864 + 255) / 256, 256, 0, stream>>>(fc1_w, wFC1, 36864);
  cvt_f16<<<(36864 + 255) / 256, 256, 0, stream>>>(fc2_w, wFC2, 36864);
  build_attn_table<<<192, 256, 0, stream>>>(rpb, attnTbl);

  const int tokBlocks = T_TOK / 8;                      // 50176, exact
  ln1_window<<<tokBlocks, 256, 0, stream>>>(x, n1w, n1b, xw);

  gemm_wmma<0><<<(6272 * 9)  / 8, 256, 0, stream>>>(xw,   wQKV,  qkv_b,  qkvb, T_TOK, 288, 96);
  attn_kernel<<<NWIN * 3, 128, 0, stream>>>(qkvb, obuf, attnTbl);
  gemm_wmma<1><<<(6272 * 3)  / 8, 256, 0, stream>>>(obuf, wPROJ, proj_b, xrw,  T_TOK, 96,  96);
  unwindow_ln2<<<tokBlocks, 256, 0, stream>>>(xrw, x, n2w, n2b, (float*)d_out, hln);
  gemm_wmma<2><<<(6272 * 12) / 8, 256, 0, stream>>>(hln,  wFC1,  fc1_b,  h1,   T_TOK, 384, 96);
  gemm_wmma<3><<<(6272 * 3)  / 8, 256, 0, stream>>>(h1,   wFC2,  fc2_b,  d_out, T_TOK, 96, 384);
}
